// Attention_87625922773171
// MI455X (gfx1250) — matrix operand order repaired and checked by reading, not compiled, not run on hardware
//
#include <hip/hip_runtime.h>

// ---------------------------------------------------------------------------
// Problem constants (from reference setup_inputs)
// ---------------------------------------------------------------------------
#define BB 4
#define CC 256
#define NTOK 5120           // 1024 z tokens + 4096 x tokens
#define NZ 1024             // z token count in input
#define NX 4096
#define LKV 1280            // 256 (z conv out) + 1024 (x conv out)
#define NH 8
#define HD 32

typedef __attribute__((ext_vector_type(16))) _Float16 v16h;
typedef __attribute__((ext_vector_type(8)))  _Float16 v8h;
typedef __attribute__((ext_vector_type(8)))  float    v8f;
typedef __attribute__((ext_vector_type(4)))  unsigned int su4;
typedef __attribute__((ext_vector_type(8)))  unsigned int su8;

static __device__ __forceinline__ v16h cmb(v8h lo, v8h hi) {
  v16h r;
#pragma unroll
  for (int i = 0; i < 8; ++i) { r[i] = lo[i]; r[i + 8] = hi[i]; }
  return r;
}

static __device__ __forceinline__ v8f wmma_f16(v16h a, v16h b, v8f c) {
  return __builtin_amdgcn_wmma_f32_16x16x32_f16(false, a, false, b, (short)0, c,
                                                false, false);
}

// ---------------------------------------------------------------------------
// Elementwise f32 -> f16 cast
// ---------------------------------------------------------------------------
__global__ void cast_f16_kernel(const float* __restrict__ in,
                                _Float16* __restrict__ out, int n) {
  int i = blockIdx.x * 256 + threadIdx.x;
  if (i < n) out[i] = (_Float16)in[i];
}

// Repack Wsr (Cout, Cin, 2, 2) -> Wc (Cout, 4*Cin) with k = (dy*2+dx)*256 + ci
__global__ void pack_wsr_kernel(const float* __restrict__ wsr,
                                _Float16* __restrict__ wc) {
  int i = blockIdx.x * 256 + threadIdx.x;
  if (i >= 256 * 1024) return;
  int co = i >> 10, r = i & 1023;
  int e = r >> 8, ci = r & 255;
  int dy = e >> 1, dx = e & 1;
  wc[i] = (_Float16)wsr[((co * 256 + ci) * 2 + dy) * 2 + dx];
}

// ---------------------------------------------------------------------------
// Generic GEMM: out(MxN) = A(MxK,f16,row-major) * W(NxK,f16,row-major)^T
// One wave computes a 16x64 tile (A fragment reused across 4 WMMAs per K step).
// K % 32 == 0, N % 64 == 0.
// ---------------------------------------------------------------------------
template <bool BIAS, bool OUTF32>
__global__ void gemm_f16_kernel(const _Float16* __restrict__ A,
                                const _Float16* __restrict__ W,
                                const float* __restrict__ bias,
                                void* __restrict__ out, int M, int N, int K) {
  int wid  = (blockIdx.x * blockDim.x + threadIdx.x) >> 5;
  int lane = threadIdx.x & 31;
  int ngroups = N >> 6;  // 64-column groups
  int mt = wid / ngroups;
  int ng = wid - mt * ngroups;
  if (mt * 16 >= M) return;  // wave-uniform
  int hi = lane >> 4, ln = lane & 15;

  const _Float16* arow = A + (size_t)(mt * 16 + ln) * K + hi * 16;
  const _Float16* w0 = W + (size_t)(ng * 64 +  0 + ln) * K + hi * 16;
  const _Float16* w1 = W + (size_t)(ng * 64 + 16 + ln) * K + hi * 16;
  const _Float16* w2 = W + (size_t)(ng * 64 + 32 + ln) * K + hi * 16;
  const _Float16* w3 = W + (size_t)(ng * 64 + 48 + ln) * K + hi * 16;

  v8f acc0 = {}, acc1 = {}, acc2 = {}, acc3 = {};
  for (int k = 0; k < K; k += 32) {
    __builtin_prefetch(arow + k + 128, 0, 1);
    v16h a  = cmb(*(const v8h*)(arow + k), *(const v8h*)(arow + k + 8));
    v16h b0 = cmb(*(const v8h*)(w0 + k), *(const v8h*)(w0 + k + 8));
    v16h b1 = cmb(*(const v8h*)(w1 + k), *(const v8h*)(w1 + k + 8));
    v16h b2 = cmb(*(const v8h*)(w2 + k), *(const v8h*)(w2 + k + 8));
    v16h b3 = cmb(*(const v8h*)(w3 + k), *(const v8h*)(w3 + k + 8));
    acc0 = wmma_f16(a, b0, acc0);
    acc1 = wmma_f16(a, b1, acc1);
    acc2 = wmma_f16(a, b2, acc2);
    acc3 = wmma_f16(a, b3, acc3);
  }

  int row0 = mt * 16 + (hi << 3);
#pragma unroll
  for (int t = 0; t < 4; ++t) {
    v8f acc = (t == 0) ? acc0 : (t == 1) ? acc1 : (t == 2) ? acc2 : acc3;
    int col = ng * 64 + t * 16 + ln;
    float bv = BIAS ? bias[col] : 0.0f;
    if (OUTF32) {
      float* o = (float*)out;
#pragma unroll
      for (int v = 0; v < 8; ++v) o[(size_t)(row0 + v) * N + col] = acc[v] + bv;
    } else {
      _Float16* o = (_Float16*)out;
#pragma unroll
      for (int v = 0; v < 8; ++v)
        o[(size_t)(row0 + v) * N + col] = (_Float16)(acc[v] + bv);
    }
  }
}

// ---------------------------------------------------------------------------
// SR conv (2x2 stride 2 as GEMM, K=1024) + bias + LayerNorm -> cat (f16)
// Block: 256 threads = 8 waves; block computes 16 output tokens x 256 chans.
// Wave w handles n-tiles {2w, 2w+1}. Grid: B * (16 z-blocks + 64 x-blocks).
// ---------------------------------------------------------------------------
__global__ void conv_ln_kernel(const _Float16* __restrict__ xh,
                               const _Float16* __restrict__ wc,
                               const float* __restrict__ bsr,
                               const float* __restrict__ gamma,
                               const float* __restrict__ beta,
                               _Float16* __restrict__ cath) {
  __shared__ float tile[16][257];
  int bi = blockIdx.x;
  int b = bi / 80, r = bi % 80;
  int inW, outW, tokbase, catoff, t0;
  if (r < 16) { inW = 32; outW = 16; tokbase = 0;    catoff = 0;   t0 = r * 16; }
  else        { inW = 64; outW = 32; tokbase = 1024; catoff = 256; t0 = (r - 16) * 16; }

  int lane = threadIdx.x & 31, w = threadIdx.x >> 5;
  int hi = lane >> 4, ln = lane & 15;

  int m  = t0 + ln;                 // output token for this A row
  int oy = m / outW, ox = m - oy * outW;

  int nt0 = w * 2, nt1 = w * 2 + 1;
  const _Float16* w0 = wc + (size_t)(nt0 * 16 + ln) * 1024 + hi * 16;
  const _Float16* w1 = wc + (size_t)(nt1 * 16 + ln) * 1024 + hi * 16;

  v8f acc0 = {}, acc1 = {};
  for (int kk = 0; kk < 32; ++kk) {
    int e = kk >> 3, c0 = (kk & 7) * 32;
    int dy = e >> 1, dx = e & 1;
    int intok = tokbase + (2 * oy + dy) * inW + (2 * ox + dx);
    const _Float16* ar = xh + ((size_t)(b * NTOK + intok) * CC + c0 + hi * 16);
    v16h a  = cmb(*(const v8h*)ar, *(const v8h*)(ar + 8));
    v16h b0 = cmb(*(const v8h*)(w0 + kk * 32), *(const v8h*)(w0 + kk * 32 + 8));
    v16h b1 = cmb(*(const v8h*)(w1 + kk * 32), *(const v8h*)(w1 + kk * 32 + 8));
    acc0 = wmma_f16(a, b0, acc0);
    acc1 = wmma_f16(a, b1, acc1);
  }

  int rowt = hi << 3;
  int c0n = nt0 * 16 + ln, c1n = nt1 * 16 + ln;
  float bs0 = bsr[c0n], bs1 = bsr[c1n];
#pragma unroll
  for (int v = 0; v < 8; ++v) {
    tile[rowt + v][c0n] = acc0[v] + bs0;
    tile[rowt + v][c1n] = acc1[v] + bs1;
  }
  __syncthreads();

  // LayerNorm: 16 threads per token
  int tok = threadIdx.x >> 4, lg = threadIdx.x & 15;
  float s = 0.f, ss = 0.f;
#pragma unroll
  for (int j = 0; j < 16; ++j) {
    float vv = tile[tok][lg + 16 * j];
    s += vv; ss += vv * vv;
  }
#pragma unroll
  for (int off = 1; off < 16; off <<= 1) {
    s  += __shfl_xor(s,  off, 32);
    ss += __shfl_xor(ss, off, 32);
  }
  float mu   = s * (1.0f / 256.0f);
  float var  = ss * (1.0f / 256.0f) - mu * mu;
  float rstd = rsqrtf(var + 1e-5f);

  _Float16* orow = cath + (size_t)(b * LKV + catoff + t0 + tok) * CC;
#pragma unroll
  for (int j = 0; j < 16; ++j) {
    int c = lg + 16 * j;
    float vv = tile[tok][c];
    orow[c] = (_Float16)((vv - mu) * rstd * gamma[c] + beta[c]);
  }
}

// ---------------------------------------------------------------------------
// Flash-style attention. One wave = (b, h, 16-query tile); kv chunks of 32.
// V chunks (32 keys x 32 dims, wave-uniform tile) are staged global->LDS by
// the Tensor Data Mover, issued at the top of each chunk and overlapped with
// the K loads + S WMMAs + softmax; a single s_wait_tensorcnt 0 precedes use.
// qh: (B, NTOK, C) f16 (c = h*32+d); kvh: (B, LKV, 512) f16 (k at c, v at 256+c)
// oh: (B, NTOK, C) f16. Lk % 32 == 0, all waves barrier-uniform.
// ---------------------------------------------------------------------------
__global__ void attn_kernel(const _Float16* __restrict__ qh,
                            const _Float16* __restrict__ kvh,
                            _Float16* __restrict__ oh,
                            int Lq, int Lk, int qoff, float scale) {
  __shared__ _Float16 Pl[8][16][40];   // P tile per wave (row-major, padded)
  __shared__ _Float16 Vl[8][32][32];   // V chunk per wave (key-major, TDM dest)

  int wid  = (blockIdx.x * blockDim.x + threadIdx.x) >> 5;
  int lane = threadIdx.x & 31;
  int wb   = threadIdx.x >> 5;
  int qtiles = Lq >> 4;
  int qt = wid % qtiles;
  int bh = wid / qtiles;
  int h = bh & 7, b = bh >> 3;
  int hi = lane >> 4, ln = lane & 15;

  // Q fragment (A, 16x32) -- loaded once
  const _Float16* qp =
      qh + ((size_t)(b * NTOK + qoff + qt * 16 + ln) * CC + h * HD + hi * 16);
  v16h aq = cmb(*(const v8h*)qp, *(const v8h*)(qp + 8));

  v8f O0 = {}, O1 = {};
  float mrow[8], lrow[8];
#pragma unroll
  for (int v = 0; v < 8; ++v) { mrow[v] = -1e30f; lrow[v] = 0.f; }

  unsigned ldsoff = (unsigned)(uintptr_t)(&Vl[wb][0][0]);

  int nc = Lk >> 5;
  for (int ck = 0; ck < nc; ++ck) {
    int k0 = ck * 32;

    // --- Issue TDM: V chunk (32 keys x 32 dims, 2B elems, row stride 512) ---
    {
      const _Float16* vbase =
          kvh + ((size_t)(b * LKV + k0) * 512 + 256 + h * HD);
      unsigned long long ga = (unsigned long long)(uintptr_t)vbase;
      su4 g0;
      g0[0] = 1u;                                  // count=1, no gather
      g0[1] = ldsoff;                              // lds_addr (bytes)
      g0[2] = (unsigned)(ga & 0xffffffffu);        // global_addr[31:0]
      g0[3] = (unsigned)((ga >> 32) & 0x01ffffffu) // global_addr[56:32]
              | 0x80000000u;                       // type=2 ("image")
      su8 g1;
      g1[0] = 0x00010000u;   // workgroup_mask=0, data_size=1 (2 bytes)
      g1[1] = (32u << 16);   // tensor_dim0 = 32  (bits 79:48)
      g1[2] = (32u << 16);   // tensor_dim1 = 32  (bits 111:80)
      g1[3] = (32u << 16);   // tile_dim0  = 32  (bits 127:112)
      g1[4] = 32u;           // tile_dim1  = 32  (bits 143:128), tile_dim2=0
      g1[5] = 512u;          // tensor_dim0_stride = 512 elems (bits 207:160)
      g1[6] = 0u;            // stride hi, tensor_dim1_stride lo
      g1[7] = 0u;
      asm volatile("tensor_load_to_lds %0, %1" ::"s"(g0), "s"(g1) : "memory");
    }

    // K fragments (B, 32x16) straight from global: rows of kv are d-contiguous
    const _Float16* kp0 =
        kvh + ((size_t)(b * LKV + k0 + ln) * 512 + h * HD + hi * 16);
    const _Float16* kp1 =
        kvh + ((size_t)(b * LKV + k0 + 16 + ln) * 512 + h * HD + hi * 16);
    v16h bk0 = cmb(*(const v8h*)kp0, *(const v8h*)(kp0 + 8));
    v16h bk1 = cmb(*(const v8h*)kp1, *(const v8h*)(kp1 + 8));

    // S = Q @ K^T for 32 keys (two 16x16 tiles)
    v8f z = {};
    v8f s0 = wmma_f16(aq, bk0, z);
    v8f s1 = wmma_f16(aq, bk1, z);

    // Online softmax update (row-wise over 16 lanes per half-wave)
#pragma unroll
    for (int v = 0; v < 8; ++v) {
      float a0 = s0[v] * scale, a1 = s1[v] * scale;
      float cm = fmaxf(a0, a1);
#pragma unroll
      for (int off = 1; off < 16; off <<= 1) cm = fmaxf(cm, __shfl_xor(cm, off, 32));
      float mn = fmaxf(mrow[v], cm);
      float p0 = __expf(a0 - mn);
      float p1 = __expf(a1 - mn);
      float rs = p0 + p1;
#pragma unroll
      for (int off = 1; off < 16; off <<= 1) rs += __shfl_xor(rs, off, 32);
      float alpha = __expf(mrow[v] - mn);
      mrow[v] = mn;
      lrow[v] = lrow[v] * alpha + rs;
      O0[v] *= alpha;
      O1[v] *= alpha;
      int rr = v + (hi << 3);
      Pl[wb][rr][ln]      = (_Float16)p0;
      Pl[wb][rr][16 + ln] = (_Float16)p1;
    }
    __syncthreads();

    // P fragment (A, 16x32) from LDS
    const _Float16* pp = &Pl[wb][ln][hi * 16];
    v16h ap = cmb(*(const v8h*)pp, *(const v8h*)(pp + 8));

    // Wait for the TDM copy of this chunk's V tile, then gather B fragments
    __builtin_amdgcn_s_wait_tensorcnt(0);
    v16h bv0, bv1;
#pragma unroll
    for (int j = 0; j < 16; ++j) {
      bv0[j] = Vl[wb][(hi << 4) + j][ln];
      bv1[j] = Vl[wb][(hi << 4) + j][16 + ln];
    }
    O0 = wmma_f16(ap, bv0, O0);
    O1 = wmma_f16(ap, bv1, O1);
    __syncthreads();
  }

  // Normalize and write out (c = h*32 + dim)
#pragma unroll
  for (int v = 0; v < 8; ++v) {
    float inv = 1.0f / lrow[v];
    int tok = qoff + qt * 16 + v + (hi << 3);
    size_t base = (size_t)(b * NTOK + tok) * CC + h * HD;
    oh[base + ln]      = (_Float16)(O0[v] * inv);
    oh[base + 16 + ln] = (_Float16)(O1[v] * inv);
  }
}

// ---------------------------------------------------------------------------
// Host launcher
// ---------------------------------------------------------------------------
extern "C" void kernel_launch(void* const* d_in, const int* in_sizes, int n_in,
                              void* d_out, int out_size, void* d_ws,
                              size_t ws_size, hipStream_t stream) {
  const float* x     = (const float*)d_in[0];
  const float* Wq    = (const float*)d_in[1];
  const float* Wkv   = (const float*)d_in[2];
  const float* Wsr   = (const float*)d_in[3];
  const float* bsr   = (const float*)d_in[4];
  const float* gamma = (const float*)d_in[5];
  const float* beta  = (const float*)d_in[6];
  const float* Wproj = (const float*)d_in[7];
  const float* bproj = (const float*)d_in[8];
  float* out = (float*)d_out;

  char* ws = (char*)d_ws;
  size_t off = 0;
  auto carve = [&](size_t bytes) -> char* {
    char* p = ws + off;
    off += (bytes + 255) & ~(size_t)255;
    return p;
  };
  _Float16* xh    = (_Float16*)carve((size_t)BB * NTOK * CC * 2);
  _Float16* qh    = (_Float16*)carve((size_t)BB * NTOK * CC * 2);
  _Float16* Wqh   = (_Float16*)carve((size_t)CC * CC * 2);
  _Float16* Wkvh  = (_Float16*)carve((size_t)2 * CC * CC * 2);
  _Float16* Wprjh = (_Float16*)carve((size_t)CC * CC * 2);
  _Float16* Wch   = (_Float16*)carve((size_t)CC * 4 * CC * 2);
  _Float16* cath  = (_Float16*)carve((size_t)BB * LKV * CC * 2);
  _Float16* kvh   = (_Float16*)carve((size_t)BB * LKV * 2 * CC * 2);
  _Float16* attnh = (_Float16*)carve((size_t)BB * NTOK * CC * 2);

  const float scale = 0.17677669529663687f;  // 1/sqrt(32)

  // 1) f32 -> f16 casts
  int nx = BB * NTOK * CC;
  cast_f16_kernel<<<(nx + 255) / 256, 256, 0, stream>>>(x, xh, nx);
  cast_f16_kernel<<<(CC * CC + 255) / 256, 256, 0, stream>>>(Wq, Wqh, CC * CC);
  cast_f16_kernel<<<(2 * CC * CC + 255) / 256, 256, 0, stream>>>(Wkv, Wkvh, 2 * CC * CC);
  cast_f16_kernel<<<(CC * CC + 255) / 256, 256, 0, stream>>>(Wproj, Wprjh, CC * CC);
  pack_wsr_kernel<<<(256 * 1024 + 255) / 256, 256, 0, stream>>>(Wsr, Wch);

  // 2) Q = x @ Wq^T   (M=20480, N=256, K=256) -> f16
  {
    int M = BB * NTOK, N = CC, K = CC;
    int waves = (M / 16) * (N / 64);
    gemm_f16_kernel<false, false><<<waves / 8, 256, 0, stream>>>(
        xh, Wqh, nullptr, qh, M, N, K);
  }

  // 3) SR conv + LN -> cat (f16).  B * (16 z + 64 x) blocks
  conv_ln_kernel<<<BB * 80, 256, 0, stream>>>(xh, Wch, bsr, gamma, beta, cath);

  // 4) KV = cat @ Wkv^T  (M=5120, N=512, K=256) -> f16
  {
    int M = BB * LKV, N = 2 * CC, K = CC;
    int waves = (M / 16) * (N / 64);
    gemm_f16_kernel<false, false><<<waves / 8, 256, 0, stream>>>(
        cath, Wkvh, nullptr, kvh, M, N, K);
  }

  // 5) Attention: z queries (1024, kv=256), x queries (4096, kv=1280)
  {
    int waves_z = BB * NH * (NZ / 16);
    attn_kernel<<<waves_z / 8, 256, 0, stream>>>(qh, kvh, attnh, NZ, 256, 0, scale);
    int waves_x = BB * NH * (NX / 16);
    attn_kernel<<<waves_x / 8, 256, 0, stream>>>(qh, kvh, attnh, NX, LKV, NZ, scale);
  }

  // 6) out = attn @ Wproj^T + bproj  (f32 to d_out)
  {
    int M = BB * NTOK, N = CC, K = CC;
    int waves = (M / 16) * (N / 64);
    gemm_f16_kernel<true, true><<<waves / 8, 256, 0, stream>>>(
        attnh, Wprjh, bproj, out, M, N, K);
  }
}
